// MBSMultiHeadAttention_82240033784478
// MI455X (gfx1250) — matrix operand order repaired and checked by reading, not compiled, not run on hardware
//
#include <hip/hip_runtime.h>
#include <hip/hip_bf16.h>
#include <cstdint>

#define EMBED 256
#define HEADS 8
#define HD 32
#define NB 3
#define CB 10          // NB*NB+1 pair buckets
#define NBUCK 32
#define BSZ 4
#define SEQ 512
#define EPS 1e-12f

typedef __attribute__((ext_vector_type(16))) __bf16 v16bf;
typedef __attribute__((ext_vector_type(8)))  float  v8f;

union FragU { unsigned u[8]; v16bf v; };

// ---- WMMA helpers (gfx1250, wave32) -------------------------------------

__device__ inline v8f wmma_bf16(v16bf a, v16bf b, v8f c) {
  // D = A(16x32 bf16) * B(32x16 bf16) + C(16x16 f32)
  return __builtin_amdgcn_wmma_f32_16x16x32_bf16(
      false, a, false, b, (short)0, c, false, false);
}

// A fragment: 16x32 bf16, ISA layout: lane L holds row M=L&15;
// lanes 0-15: K pairs {0..7,16..23}, lanes 16-31: {8..15,24..31}.
// `row` points at 32 contiguous bf16 (16 dwords) for this lane's row.
__device__ inline v16bf load_fragA(const unsigned* row, int half) {
  FragU f;
#pragma unroll
  for (int r = 0; r < 8; ++r)
    f.u[r] = row[(r & 3) + ((r >> 2) << 3) + (half << 2)];
  return f.v;
}

// B fragment: 32x16 bf16.  Lane L holds column N=L&15; lanes 0-15 carry
// K {0..7,16..23}, lanes 16-31 carry K {8..15,24..31} (same as A), i.e. we read the
// transposed matrix row-major.  `row` = 32 contiguous bf16 of B^T row N.
__device__ inline v16bf load_fragB(const unsigned* row, int half) {
  FragU f;
#pragma unroll
  for (int r = 0; r < 8; ++r)
    f.u[r] = row[(r & 3) + ((r >> 2) << 3) + (half << 2)];
  return f.v;
}

__device__ inline int bm_fun(int tq, int tk) {
  return (tq != 0 && tk != 0) ? (tq - 1) * NB + tk : 0;
}

// ---- K0: W1_/W2_ = softmax(alpha, axis=b) - weighted mix -----------------

__global__ __launch_bounds__(256) void mix_kernel(
    const float* __restrict__ W1, const float* __restrict__ a1,
    const float* __restrict__ W2, const float* __restrict__ a2,
    float* __restrict__ W1m, float* __restrict__ W2m) {
  const int c = blockIdx.x >> 3, h = blockIdx.x & 7;
  float x0 = a1[(c * NB + 0) * HEADS + h];
  float x1 = a1[(c * NB + 1) * HEADS + h];
  float x2 = a1[(c * NB + 2) * HEADS + h];
  float mx = fmaxf(x0, fmaxf(x1, x2));
  float e0 = __expf(x0 - mx), e1 = __expf(x1 - mx), e2 = __expf(x2 - mx);
  float inv = 1.f / (e0 + e1 + e2);
  e0 *= inv; e1 *= inv; e2 *= inv;
  float y0 = a2[(c * NB + 0) * HEADS + h];
  float y1 = a2[(c * NB + 1) * HEADS + h];
  float y2 = a2[(c * NB + 2) * HEADS + h];
  float my = fmaxf(y0, fmaxf(y1, y2));
  float f0 = __expf(y0 - my), f1 = __expf(y1 - my), f2 = __expf(y2 - my);
  float jnv = 1.f / (f0 + f1 + f2);
  f0 *= jnv; f1 *= jnv; f2 *= jnv;
  for (int e = threadIdx.x; e < HD * HD; e += 256) {
    size_t o = ((size_t)c * HEADS + h) * (HD * HD) + e;
    size_t s0 = ((size_t)0 * HEADS + h) * (HD * HD) + e;
    size_t s1 = ((size_t)1 * HEADS + h) * (HD * HD) + e;
    size_t s2 = ((size_t)2 * HEADS + h) * (HD * HD) + e;
    W1m[o] = e0 * W1[s0] + e1 * W1[s1] + e2 * W1[s2];
    W2m[o] = f0 * W2[s0] + f1 * W2[s1] + f2 * W2[s2];
  }
}

// ---- K1: T5 relative-position bucket table ------------------------------

__global__ __launch_bounds__(256) void bucket_kernel(int* __restrict__ bucket) {
  int idx = blockIdx.x * 256 + threadIdx.x;
  if (idx >= SEQ * SEQ) return;
  int q = idx >> 9, k = idx & (SEQ - 1);
  int n = q - k;                       // n = -(k - q)
  int ret = (n < 0) ? 16 : 0;
  n = n < 0 ? -n : n;
  int bkt;
  if (n < 8) {
    bkt = n;
  } else {
    // 8 + floor(log(n/8)/log(40/8) * 8), clamped to 15
    float vl = logf((float)n * 0.125f) * (8.0f / 1.6094379124341003f);
    int v = 8 + (int)vl;
    bkt = v > 15 ? 15 : v;
  }
  bucket[idx] = ret + bkt;
}

// ---- K2: fused type-selected QKV projection + qhat/vhat precompute ------
// One block per token.  q/k/v = x . W{q,k,v}[ts] ; then
//   qhat[b,h,t,l,n] = sum_m q[h,m] * W1m[bm(t0,t),h,m,n]        (bf16)
//   vhatT[b,h,t,m,l] = sum_n v[h,n] * W2m[bm(t,t0),h,n,m]       (bf16)
//   k stored as bf16 [b,h,l,d]

__global__ __launch_bounds__(256) void proj_kernel(
    const float* __restrict__ x, const int* __restrict__ ts,
    const float* __restrict__ Wq, const float* __restrict__ Wk,
    const float* __restrict__ Wv, const float* __restrict__ W1m,
    const float* __restrict__ W2m, __bf16* __restrict__ kb,
    __bf16* __restrict__ qhat, __bf16* __restrict__ vhatT) {
  __shared__ float xs[EMBED], qs[EMBED], vs[EMBED];
  const int b = blockIdx.x / SEQ, l = blockIdx.x % SEQ;
  const int i = threadIdx.x;
  xs[i] = x[((size_t)b * SEQ + l) * EMBED + i];
  __syncthreads();
  const int t0 = ts[b * SEQ + l];
  const float* wq = Wq + (size_t)t0 * EMBED * EMBED + i;
  const float* wk = Wk + (size_t)t0 * EMBED * EMBED + i;
  const float* wv = Wv + (size_t)t0 * EMBED * EMBED + i;
  float q = 0.f, k = 0.f, v = 0.f;
#pragma unroll 8
  for (int m = 0; m < EMBED; ++m) {
    float xv = xs[m];
    q += xv * wq[(size_t)m * EMBED];
    k += xv * wk[(size_t)m * EMBED];
    v += xv * wv[(size_t)m * EMBED];
  }
  const int h = i >> 5, j = i & 31;
  kb[(((size_t)(b * HEADS + h)) * SEQ + l) * HD + j] = (__bf16)k;
  qs[i] = q; vs[i] = v;
  __syncthreads();
#pragma unroll
  for (int t = 0; t < 4; ++t) {
    const int c1 = bm_fun(t0, t);           // candidate key type = t
    const float* w1 = W1m + ((size_t)c1 * HEADS + h) * (HD * HD) + j;
    float aq = 0.f;
#pragma unroll
    for (int m = 0; m < HD; ++m) aq += qs[h * HD + m] * w1[m * HD];
    qhat[((((size_t)(b * HEADS + h)) * 4 + t) * SEQ + l) * HD + j] = (__bf16)aq;

    const int c2 = bm_fun(t, t0);           // candidate query type = t
    const float* w2 = W2m + ((size_t)c2 * HEADS + h) * (HD * HD) + j;
    float av = 0.f;
#pragma unroll
    for (int n = 0; n < HD; ++n) av += vs[h * HD + n] * w2[n * HD];
    vhatT[((((size_t)(b * HEADS + h)) * 4 + t) * HD + j) * SEQ + l] = (__bf16)av;
  }
}

// ---- K3: flash attention with pair-bucket select, WMMA bf16 -------------
// Block = 256 threads = 8 waves; each wave owns 16 query rows.
// Per 32-wide K chunk: 8 score WMMAs (4 type variants x 2 column halves),
// per-lane column select, online softmax, P staged via LDS into A layout,
// 8 PV WMMAs into 4 per-row-type accumulator sets.

__global__ __launch_bounds__(256) void attn_kernel(
    const __bf16* __restrict__ qhat, const __bf16* __restrict__ kb,
    const __bf16* __restrict__ vhatT, const int* __restrict__ ts,
    const int* __restrict__ bucket, const float* __restrict__ rp,
    const float* __restrict__ mask, float* __restrict__ ctx) {
  __shared__ __bf16 pT[8][16][32];   // per-wave P staging tile
  const int tid  = threadIdx.x;
  const int w    = tid >> 5, lane = tid & 31;
  const int half = lane >> 4, lr = lane & 15;
  const int b = blockIdx.z, h = blockIdx.y;
  const int qbase = blockIdx.x * 128 + w * 16;
  const int bh = b * HEADS + h;
  const float scale = 0.17677669529663687f;  // 1/sqrt(HD)

  // resident qhat A-fragments, one per candidate key type
  v16bf qf[4];
#pragma unroll
  for (int t = 0; t < 4; ++t) {
    const unsigned* row = (const unsigned*)(
        qhat + ((((size_t)bh * 4) + t) * SEQ + (qbase + lr)) * HD);
    qf[t] = load_fragA(row, half);
  }

  float mrow[8], lrow[8];
  int tqr[8];
#pragma unroll
  for (int i = 0; i < 8; ++i) {
    mrow[i] = -INFINITY; lrow[i] = 0.f;
    tqr[i] = ts[b * SEQ + qbase + i + 8 * half];
  }

  v8f acc[4][2];
#pragma unroll
  for (int t = 0; t < 4; ++t) { acc[t][0] = {}; acc[t][1] = {}; }

  for (int kc = 0; kc < SEQ / 32; ++kc) {
    const int kbase = kc * 32;
    float vals[2][8], pv[2][8], vmax[8];
#pragma unroll
    for (int i = 0; i < 8; ++i) vmax[i] = -INFINITY;

#pragma unroll
    for (int hn = 0; hn < 2; ++hn) {
      const int kcol = kbase + hn * 16 + lr;   // this lane's key column
      const unsigned* krow =
          (const unsigned*)(kb + ((size_t)bh * SEQ + kcol) * HD);
      v16bf bk = load_fragB(krow, half);
      v8f z = {};
      v8f s0 = wmma_bf16(qf[0], bk, z);
      v8f s1 = wmma_bf16(qf[1], bk, z);
      v8f s2 = wmma_bf16(qf[2], bk, z);
      v8f s3 = wmma_bf16(qf[3], bk, z);
      const int tk = ts[b * SEQ + kcol];
      v8f s = (tk == 1) ? s1 : (tk == 2) ? s2 : (tk == 3) ? s3 : s0;
#pragma unroll
      for (int i = 0; i < 8; ++i) {
        const int Q = qbase + i + 8 * half;
        float vsc = s[i] * scale;
        const int c = bm_fun(tqr[i], tk);
        const int bk2 = bucket[Q * SEQ + kcol];
        vsc += rp[(c * NBUCK + bk2) * HEADS + h];
        vsc += mask[((size_t)b * SEQ + Q) * SEQ + kcol];
        vals[hn][i] = vsc;
        vmax[i] = fmaxf(vmax[i], vsc);
      }
    }
    // row max across the 16-lane column group
#pragma unroll
    for (int i = 0; i < 8; ++i) {
#pragma unroll
      for (int off = 1; off < 16; off <<= 1)
        vmax[i] = fmaxf(vmax[i], __shfl_xor(vmax[i], off, 32));
    }
    // online softmax update
#pragma unroll
    for (int i = 0; i < 8; ++i) {
      float mnew = fmaxf(mrow[i], vmax[i]);
      float corr = __expf(mrow[i] - mnew);
      mrow[i] = mnew;
      float p0 = __expf(vals[0][i] - mnew);
      float p1 = __expf(vals[1][i] - mnew);
      float rs = p0 + p1;
#pragma unroll
      for (int off = 1; off < 16; off <<= 1) rs += __shfl_xor(rs, off, 32);
      lrow[i] = lrow[i] * corr + rs;
#pragma unroll
      for (int t = 0; t < 4; ++t) {
        acc[t][0][i] *= corr;
        acc[t][1][i] *= corr;
      }
      pv[0][i] = p0; pv[1][i] = p1;
    }
    // stage P (C/D layout -> row-major) into this wave's LDS tile
#pragma unroll
    for (int hn = 0; hn < 2; ++hn)
#pragma unroll
      for (int i = 0; i < 8; ++i)
        pT[w][i + 8 * half][hn * 16 + lr] = (__bf16)pv[hn][i];
    __syncthreads();
    v16bf pf = load_fragA((const unsigned*)&pT[w][lr][0], half);
    // PV: 4 type variants x 2 output halves, K = 32 chunk
#pragma unroll
    for (int t = 0; t < 4; ++t) {
#pragma unroll
      for (int oh = 0; oh < 2; ++oh) {
        const unsigned* vrow = (const unsigned*)(
            vhatT + (((size_t)bh * 4 + t) * HD + (oh * 16 + lr)) * SEQ + kbase);
        v16bf bv = load_fragB(vrow, half);
        acc[t][oh] = wmma_bf16(pf, bv, acc[t][oh]);
      }
    }
    __syncthreads();
  }
  // epilogue: per-row type select, normalize, write ctx [B, L, E]
#pragma unroll
  for (int oh = 0; oh < 2; ++oh) {
#pragma unroll
    for (int i = 0; i < 8; ++i) {
      const int t = tqr[i];
      float a = (t == 1) ? acc[1][oh][i]
              : (t == 2) ? acc[2][oh][i]
              : (t == 3) ? acc[3][oh][i]
                         : acc[0][oh][i];
      const int Q = qbase + i + 8 * half;
      ctx[((size_t)b * SEQ + Q) * EMBED + h * HD + oh * 16 + lr] =
          a / lrow[i];
    }
  }
}

// ---- K4: residual + LayerNorm -------------------------------------------

__global__ __launch_bounds__(256) void ln_kernel(
    const float* __restrict__ x, const float* __restrict__ ctx,
    const float* __restrict__ gamma, const float* __restrict__ beta,
    float* __restrict__ out) {
  __shared__ float red[256];
  const int bl = blockIdx.x, i = threadIdx.x;
  float xi = ctx[(size_t)bl * EMBED + i] + x[(size_t)bl * EMBED + i];
  red[i] = xi;
  __syncthreads();
  for (int s = 128; s > 0; s >>= 1) {
    if (i < s) red[i] += red[i + s];
    __syncthreads();
  }
  float mu = red[0] * (1.0f / EMBED);
  __syncthreads();
  float d = xi - mu;
  red[i] = d * d;
  __syncthreads();
  for (int s = 128; s > 0; s >>= 1) {
    if (i < s) red[i] += red[i + s];
    __syncthreads();
  }
  float var = red[0] * (1.0f / EMBED);
  out[(size_t)bl * EMBED + i] = d * rsqrtf(var + EPS) * gamma[i] + beta[i];
}

// ---- launcher ------------------------------------------------------------

extern "C" void kernel_launch(void* const* d_in, const int* in_sizes, int n_in,
                              void* d_out, int out_size, void* d_ws,
                              size_t ws_size, hipStream_t stream) {
  const float* x     = (const float*)d_in[0];
  const float* mask  = (const float*)d_in[1];
  const int*   ts    = (const int*)d_in[2];
  const float* Wq    = (const float*)d_in[3];
  const float* Wk    = (const float*)d_in[4];
  const float* Wv    = (const float*)d_in[5];
  const float* W1    = (const float*)d_in[6];
  const float* a1    = (const float*)d_in[7];
  const float* W2    = (const float*)d_in[8];
  const float* a2    = (const float*)d_in[9];
  const float* rp    = (const float*)d_in[10];
  const float* gamma = (const float*)d_in[11];
  const float* beta  = (const float*)d_in[12];
  float* out = (float*)d_out;

  // workspace layout (~12.7 MB total)
  char* ws = (char*)d_ws;
  size_t off = 0;
  float* W1m = (float*)(ws + off); off += (size_t)CB * HEADS * HD * HD * 4;
  float* W2m = (float*)(ws + off); off += (size_t)CB * HEADS * HD * HD * 4;
  int* bucket = (int*)(ws + off);  off += (size_t)SEQ * SEQ * 4;
  __bf16* kb    = (__bf16*)(ws + off); off += (size_t)BSZ * HEADS * SEQ * HD * 2;
  __bf16* qhat  = (__bf16*)(ws + off); off += (size_t)BSZ * HEADS * 4 * SEQ * HD * 2;
  __bf16* vhatT = (__bf16*)(ws + off); off += (size_t)BSZ * HEADS * 4 * HD * SEQ * 2;
  float* ctx = (float*)(ws + off);     off += (size_t)BSZ * SEQ * EMBED * 4;

  mix_kernel<<<CB * HEADS, 256, 0, stream>>>(W1, a1, W2, a2, W1m, W2m);
  bucket_kernel<<<(SEQ * SEQ) / 256, 256, 0, stream>>>(bucket);
  proj_kernel<<<BSZ * SEQ, 256, 0, stream>>>(x, ts, Wq, Wk, Wv, W1m, W2m,
                                             kb, qhat, vhatT);
  attn_kernel<<<dim3(SEQ / 128, HEADS, BSZ), 256, 0, stream>>>(
      qhat, kb, vhatT, ts, bucket, rp, mask, ctx);
  ln_kernel<<<BSZ * SEQ, 256, 0, stream>>>(x, ctx, gamma, beta, out);
}
